// GSU_25228637897431
// MI455X (gfx1250) — matrix operand order repaired and checked by reading, not compiled, not run on hardware
//
#include <hip/hip_runtime.h>
#include <hip/hip_bf16.h>
#include <math.h>

// ---------------------------------------------------------------------------
// MI455X / gfx1250 fused two-branch attention-pooling kernel.
// bf16 WMMA (V_WMMA_F32_16X16X32_BF16) for both GEMMs, fp32 accumulate.
// X tiles staged into LDS by the Tensor Data Mover (tensor_load_to_lds with
// D# row padding), drained with s_wait_tensorcnt, overlapped with compute.
// Both branches folded into one 512-workgroup launch (blockIdx.y = branch).
// ---------------------------------------------------------------------------

typedef __attribute__((ext_vector_type(16))) __bf16 v16bf;
typedef __attribute__((ext_vector_type(8)))  float  v8f;
typedef __attribute__((ext_vector_type(4)))  unsigned int u32x4;
typedef __attribute__((ext_vector_type(8)))  int  i32x8;
typedef __attribute__((ext_vector_type(4)))  int  i32x4;

union BF16x16 { v16bf v; unsigned short u[16]; __bf16 b[16]; };

__device__ __forceinline__ unsigned short f2bfu(float f) {
  union { __bf16 b; unsigned short u; } c;
  c.b = (__bf16)f;               // native f32->bf16 convert (v_cvt_pk_bf16_f32)
  return c.u;
}

#define TS 64          // S rows per tile
#define NB 256         // batch
#define NS 1024        // sequence
#define ND 64          // doc emb
#define HD 128         // H*D
#define KF 312         // feat width (4*64 + 56)
#define XSTR 65        // LDS row stride (TDM pad: +1 dword per 64-dword row)

#define HAVE_TDM __has_builtin(__builtin_amdgcn_tensor_load_to_lds)

__device__ __forceinline__ void wait_tensorcnt0() {
#if __has_builtin(__builtin_amdgcn_s_wait_tensorcnt)
  __builtin_amdgcn_s_wait_tensorcnt(0);
#else
  asm volatile("s_wait_tensorcnt 0x0" ::: "memory");
#endif
}

#if HAVE_TDM
// DMA a 64x64 fp32 tile (row stride 64 elems in memory) into LDS at ldsoff,
// padding LDS rows to 65 dwords (pad_interval=64dw, pad_amount=1dw).
__device__ __forceinline__ void tdm_load_tile(const float* gsrc,
                                              unsigned int ldsoff) {
  const unsigned long long ga = (unsigned long long)(uintptr_t)gsrc;
  u32x4 g0;
  g0[0] = 1u;                                    // count=1, user descriptor
  g0[1] = ldsoff;                                // lds_addr (bytes)
  g0[2] = (unsigned int)ga;                      // global_addr[31:0]
  g0[3] = (unsigned int)((ga >> 32) & 0x01FFFFFFu) | (2u << 30); // [56:32]|type=2
  i32x8 g1;
  g1[0] = 0x01520000;   // data_size=4B | pad_enable | pad_interval=64dw | pad=1dw
  g1[1] = 64 << 16;     // tensor_dim0 = 64
  g1[2] = 64 << 16;     // tensor_dim1 = 64
  g1[3] = 64 << 16;     // tile_dim0   = 64
  g1[4] = 64;           // tile_dim1   = 64, tile_dim2 = 0
  g1[5] = 64;           // tensor_dim0_stride = 64 elements
  g1[6] = 0;
  g1[7] = 0;
  i32x4 z4 = {0, 0, 0, 0};
#if __clang_major__ >= 23
  i32x8 z8 = {0, 0, 0, 0, 0, 0, 0, 0};
  __builtin_amdgcn_tensor_load_to_lds(g0, g1, z4, z4, z8, 0);
#else
  __builtin_amdgcn_tensor_load_to_lds(g0, g1, z4, z4, 0);
#endif
}
#endif

// ---------------------------------------------------------------------------
// Prep: side-feature concat + fp32 -> bf16 weight conversion into workspace.
// ---------------------------------------------------------------------------
__global__ __launch_bounds__(256) void prep_kernel(
    const float* __restrict__ uid, const float* __restrict__ osb,
    const float* __restrict__ zipb,
    const float* __restrict__ w_lx, const float* __restrict__ w_nx,
    const float* __restrict__ lw1, const float* __restrict__ nw1,
    float* __restrict__ side,
    unsigned short* __restrict__ wlxb, unsigned short* __restrict__ wnxb,
    unsigned short* __restrict__ lw1b, unsigned short* __restrict__ nw1b) {
  int i = blockIdx.x * 256 + threadIdx.x;
  if (i < NB * 56) {
    int b = i / 56, c = i % 56;
    float v;
    if (c < 32)      v = uid[b * 32 + c];
    else if (c < 40) v = osb[b * 8 + (c - 32)];
    else             v = zipb[b * 16 + (c - 40)];
    side[i] = v;
  }
  if (i < ND * HD) { wlxb[i] = f2bfu(w_lx[i]); wnxb[i] = f2bfu(w_nx[i]); }
  if (i < KF * 40) { lw1b[i] = f2bfu(lw1[i]);  nw1b[i] = f2bfu(nw1[i]); }
}

// ---------------------------------------------------------------------------
// y_trans = l2norm(prelu(y @ w_ty + b_ty))   [256,128]  (tiny, VALU)
// ---------------------------------------------------------------------------
__global__ __launch_bounds__(128) void ytrans_kernel(
    const float* __restrict__ y, const float* __restrict__ w_ty,
    const float* __restrict__ b_ty, const float* __restrict__ a_ty,
    float* __restrict__ yT) {
  __shared__ float st[HD];
  __shared__ float yb[ND];
  __shared__ float srn;
  const int b = blockIdx.x, j = threadIdx.x;
  if (j < ND) yb[j] = y[b * ND + j];
  __syncthreads();
  float acc = b_ty[j];
  for (int k = 0; k < ND; ++k) acc += yb[k] * w_ty[k * HD + j];
  float a = a_ty[0];
  acc = (acc >= 0.0f) ? acc : a * acc;
  st[j] = acc;
  __syncthreads();
  if (j == 0) {
    float s = 0.0f;
    for (int k = 0; k < HD; ++k) s += st[k] * st[k];
    srn = 1.0f / fmaxf(sqrtf(s), 1e-12f);
  }
  __syncthreads();
  yT[b * HD + j] = st[j] * srn;
}

// ---------------------------------------------------------------------------
// Fused branch kernel: blockIdx.y selects branch, blockIdx.x = batch row.
// One workgroup (8 waves) per (branch, batch) pair, 16 S-tiles each.
// ---------------------------------------------------------------------------
__global__ __launch_bounds__(256) void fused_branch_kernel(
    const float* __restrict__ x0, const float* __restrict__ x1,   // [256,1024,64]
    const float* __restrict__ bv0, const float* __restrict__ bv1, // [128]
    const float* __restrict__ aT0, const float* __restrict__ aT1, // scalars
    const unsigned short* __restrict__ Wb0,
    const unsigned short* __restrict__ Wb1,                       // [64,128] bf16
    const unsigned short* __restrict__ w1b0,
    const unsigned short* __restrict__ w1b1,                      // [312,40] bf16
    const float* __restrict__ b1v0, const float* __restrict__ b1v1, // [40]
    const float* __restrict__ a10, const float* __restrict__ a11,   // scalars
    const float* __restrict__ w20, const float* __restrict__ w21,   // [40]
    const float* __restrict__ b20, const float* __restrict__ b21,   // [1]
    const float* __restrict__ yT,            // [256,128] y_trans
    const float* __restrict__ side,          // [256,56]
    float* __restrict__ outp)                // [2,256,128]
{
  __shared__ float sX[TS * XSTR];    // TDM-staged X tile (padded rows)
  __shared__ float sT[TS * 132];     // normalized transform tile (padded)
  __shared__ float sY[HD];
  __shared__ float sSide[56];
  __shared__ float sRn[TS];
  __shared__ float sP[128];

  const int br   = blockIdx.y;
  const int b    = blockIdx.x;
  const int tid  = threadIdx.x;
  const int wave = tid >> 5;
  const int lane = tid & 31;
  const int lr   = lane & 15;      // row / col within 16
  const int hi   = lane >> 4;      // wave half (selects K group)

  const float* __restrict__ x    = br ? x1 : x0;
  const float* __restrict__ bvec = br ? bv1 : bv0;
  const unsigned short* __restrict__ Wb  = br ? Wb1 : Wb0;
  const unsigned short* __restrict__ w1b = br ? w1b1 : w1b0;
  const float* __restrict__ b1 = br ? b1v1 : b1v0;
  const float* __restrict__ w2 = br ? w21 : w20;
  const float alphaT = br ? aT1[0] : aT0[0];
  const float a1     = br ? a11[0] : a10[0];
  const float b2     = br ? b21[0] : b20[0];

  if (tid < HD)  sY[tid]    = yT[b * HD + tid];
  if (tid < 56)  sSide[tid] = side[b * 56 + tid];

  // Per-lane GEMM2 epilogue constants (col = ctile*16 + lr)
  const float b1_0 = b1[lr];
  const float b1_1 = b1[16 + lr];
  const float b1_2 = (32 + lr < 40) ? b1[32 + lr] : 0.0f;
  const float w2_0 = w2[lr];
  const float w2_1 = w2[16 + lr];
  const float w2_2 = (32 + lr < 40) ? w2[32 + lr] : 0.0f;

  // --- GEMM1 B fragments (tile-invariant): wave owns column tile ct=wave ---
  // B 32x16 bf16 layout: col = lane&15, lanes>=16 hold K+16; elem e -> K=kb+e
  BF16x16 B0, B1;
  {
    const int col = wave * 16 + lr;
    const int kb  = hi * 16;
#pragma unroll
    for (int e = 0; e < 16; ++e) {
      B0.u[e] = Wb[(kb + e) * HD + col];
      B1.u[e] = Wb[(kb + e + 32) * HD + col];
    }
  }

  // Per-thread output accumulator (tid<128 holds out[br, b, tid])
  float outAcc = 0.0f;

  // GEMM2 per-lane row pointers (into sT, tile-invariant)
  const int frow = wave * 16 + lr;     // feat row 0..127, head-major
  const int fh   = frow >> 6;
  const int fsr  = frow & 63;
  const float* srowT = &sT[fsr * 132 + fh * 64];
  const float* syh   = &sY[fh * 64];

  const float* xbase = x + (size_t)b * NS * ND;
#if HAVE_TDM
  const unsigned int sXoff = (unsigned int)(uintptr_t)&sX[0];
  if (wave == 0) tdm_load_tile(xbase, sXoff);   // prefetch tile 0
#endif
  __syncthreads();

  for (int tile = 0; tile < NS / TS; ++tile) {
#if HAVE_TDM
    wait_tensorcnt0();                 // wave0 drains its TDM; others no-op
    __syncthreads();                   // sX visible to all waves
#else
    {                                   // cooperative fallback copy
      const float* xg = xbase + (size_t)tile * TS * ND;
      for (int i = tid; i < TS * ND; i += 256) {
        const int r = i >> 6, c = i & 63;
        sX[r * XSTR + c] = xg[i];
      }
    }
    __syncthreads();
#endif

    // ================= GEMM1: T = prelu(X @ W + b), [64 x 128] ============
#pragma unroll
    for (int rt = 0; rt < 4; ++rt) {
      // A 16x32 bf16 layout: row = lane&15; elem e -> Klocal = e + hi*16
      BF16x16 A0, A1;
      const float* xr = &sX[(rt * 16 + lr) * XSTR];
#pragma unroll
      for (int e = 0; e < 16; ++e) {
        const int kl = e + hi * 16;
        A0.b[e] = (__bf16)xr[kl];
        A1.b[e] = (__bf16)xr[kl + 32];
      }
      v8f acc = {};
      acc = __builtin_amdgcn_wmma_f32_16x16x32_bf16(false, A0.v, false, B0.v,
                                                    (short)0, acc, false, false);
      acc = __builtin_amdgcn_wmma_f32_16x16x32_bf16(false, A1.v, false, B1.v,
                                                    (short)0, acc, false, false);
      // epilogue: bias + prelu -> LDS (C/D layout: m = v + 8*hi, n = lane&15)
      const int n  = wave * 16 + lr;
      const float bn = bvec[n];
#pragma unroll
      for (int v = 0; v < 8; ++v) {
        const int m = rt * 16 + v + hi * 8;
        float t = acc[v] + bn;
        t = (t >= 0.0f) ? t : alphaT * t;
        sT[m * 132 + n] = t;
      }
    }
    __syncthreads();                   // sX reads done; sT complete

#if HAVE_TDM
    if (wave == 0 && tile + 1 < NS / TS)   // overlap next DMA with GEMM2
      tdm_load_tile(xbase + (size_t)(tile + 1) * TS * ND, sXoff);
#endif

    // ================= row l2-norm over all 128 columns ====================
    if (tid < TS) {
      float ss = 0.0f;
      for (int j = 0; j < HD; ++j) { float t = sT[tid * 132 + j]; ss += t * t; }
      sRn[tid] = 1.0f / fmaxf(sqrtf(ss), 1e-12f);
    }
    __syncthreads();
    for (int i = tid; i < TS * HD; i += 256) {
      const int r = i >> 7, c = i & 127;
      sT[r * 132 + c] *= sRn[r];
    }
    __syncthreads();

    // ===== GEMM2: h1 = prelu(feat @ w1 + b1), feat synthesized on the fly ==
    // 128 feat rows (head-major), K = 312 (pad 320), N = 40 (pad 48)
    v8f c0 = {}, c1 = {}, c2 = {};
#pragma unroll
    for (int kk = 0; kk < 10; ++kk) {
      const int kbase = kk * 32;
      BF16x16 A;
#pragma unroll
      for (int e = 0; e < 16; ++e) {
        const int kl = e + hi * 16;
        const int k  = kbase + kl;            // uniform within each lane half
        float f;
        if (k < 64)        f = srowT[k];                         // xh
        else if (k < 128)  f = srowT[k - 64] * syh[k - 64];      // xh*yh
        else if (k < 192)  f = srowT[k - 128] - syh[k - 128];    // xh-yh
        else if (k < 256)  f = syh[k - 192];                     // yh
        else if (k < KF)   f = sSide[k - 256];                   // side
        else               f = 0.0f;                             // K pad
        A.b[e] = (__bf16)f;
      }
      BF16x16 Bf0, Bf1, Bf2;
      const int kb2 = kbase + hi * 16;
#pragma unroll
      for (int e = 0; e < 16; ++e) {
        const int k = kb2 + e;
        const bool kok = (k < KF);
        Bf0.u[e] = kok ? w1b[k * 40 + lr] : (unsigned short)0;
        Bf1.u[e] = kok ? w1b[k * 40 + 16 + lr] : (unsigned short)0;
        Bf2.u[e] = (kok && (32 + lr) < 40) ? w1b[k * 40 + 32 + lr]
                                           : (unsigned short)0;   // N pad
      }
      c0 = __builtin_amdgcn_wmma_f32_16x16x32_bf16(false, A.v, false, Bf0.v,
                                                   (short)0, c0, false, false);
      c1 = __builtin_amdgcn_wmma_f32_16x16x32_bf16(false, A.v, false, Bf1.v,
                                                   (short)0, c1, false, false);
      c2 = __builtin_amdgcn_wmma_f32_16x16x32_bf16(false, A.v, false, Bf2.v,
                                                   (short)0, c2, false, false);
    }

    // ===== p[m] = b2 + sum_n prelu(h1[m,n] + b1[n]) * w2[n] ================
    // Straight from accumulators: per v, lanes of each half hold 16 cols of
    // one row -> deterministic 4-step xor-shuffle reduction within the half.
#pragma unroll
    for (int v = 0; v < 8; ++v) {
      float t0 = c0[v] + b1_0; t0 = (t0 >= 0.0f) ? t0 : a1 * t0;
      float t1 = c1[v] + b1_1; t1 = (t1 >= 0.0f) ? t1 : a1 * t1;
      float t2 = c2[v] + b1_2; t2 = (t2 >= 0.0f) ? t2 : a1 * t2;
      float s = t0 * w2_0 + t1 * w2_1 + t2 * w2_2;
      s += __shfl_xor(s, 1);
      s += __shfl_xor(s, 2);
      s += __shfl_xor(s, 4);
      s += __shfl_xor(s, 8);
      if (lr == 0) sP[wave * 16 + v + hi * 8] = s + b2;
    }
    __syncthreads();

    // ======== out[b,h,d] += sum_s p[h,s] * xh[s,d]  (register accumulate) ==
    if (tid < 128) {
      const int hh = tid >> 6, d = tid & 63;
      const float* pp = &sP[hh * 64];
      float acc = 0.0f;
      for (int r = 0; r < TS; ++r) acc += pp[r] * sT[r * 132 + hh * 64 + d];
      outAcc += acc;
    }
    __syncthreads();   // sT/sP reused next tile
  }

  if (tid < 128)
    outp[(size_t)br * NB * HD + (size_t)b * HD + tid] = outAcc;
}

// ---------------------------------------------------------------------------
// Host launcher
// ---------------------------------------------------------------------------
extern "C" void kernel_launch(void* const* d_in, const int* in_sizes, int n_in,
                              void* d_out, int out_size, void* d_ws, size_t ws_size,
                              hipStream_t stream) {
  const float* local_x    = (const float*)d_in[0];
  const float* nonlocal_x = (const float*)d_in[1];
  const float* y_embs     = (const float*)d_in[2];
  const float* uid        = (const float*)d_in[3];
  const float* osb        = (const float*)d_in[4];
  const float* zipb       = (const float*)d_in[5];
  const float* w_ty = (const float*)d_in[6];
  const float* b_ty = (const float*)d_in[7];
  const float* a_ty = (const float*)d_in[8];
  const float* w_lx = (const float*)d_in[9];
  const float* b_lx = (const float*)d_in[10];
  const float* a_lx = (const float*)d_in[11];
  const float* w_nx = (const float*)d_in[12];
  const float* b_nx = (const float*)d_in[13];
  const float* a_nx = (const float*)d_in[14];
  const float* lw1  = (const float*)d_in[15];
  const float* lb1  = (const float*)d_in[16];
  const float* la1  = (const float*)d_in[17];
  const float* lw2  = (const float*)d_in[18];
  const float* lb2  = (const float*)d_in[19];
  const float* nw1  = (const float*)d_in[20];
  const float* nb1  = (const float*)d_in[21];
  const float* na1  = (const float*)d_in[22];
  const float* nw2  = (const float*)d_in[23];
  const float* nb2  = (const float*)d_in[24];

  char* ws = (char*)d_ws;
  float*          yT   = (float*)(ws + 0);                // 256*128*4 = 131072 B
  float*          side = (float*)(ws + 131072);           // 256*56*4  =  57344 B
  unsigned short* wlxb = (unsigned short*)(ws + 188416);  // 8192*2
  unsigned short* wnxb = (unsigned short*)(ws + 204800);  // 8192*2
  unsigned short* lw1b = (unsigned short*)(ws + 221184);  // 12480*2
  unsigned short* nw1b = (unsigned short*)(ws + 246144);  // 12480*2

  float* out = (float*)d_out;

  prep_kernel<<<64, 256, 0, stream>>>(uid, osb, zipb, w_lx, w_nx, lw1, nw1,
                                      side, wlxb, wnxb, lw1b, nw1b);
  ytrans_kernel<<<NB, 128, 0, stream>>>(y_embs, w_ty, b_ty, a_ty, yT);

  dim3 grid(NB, 2);
  fused_branch_kernel<<<grid, 256, 0, stream>>>(
      local_x, nonlocal_x,
      b_lx, b_nx, a_lx, a_nx,
      wlxb, wnxb, lw1b, nw1b,
      lb1, nb1, la1, na1,
      lw2, nw2, lb2, nb2,
      yT, side, out);

  (void)in_sizes; (void)n_in; (void)out_size; (void)ws_size;
}
